// OutputLayer_80444737454618
// MI455X (gfx1250) — matrix operand order repaired and checked by reading, not compiled, not run on hardware
//
#include <hip/hip_runtime.h>
#include <hip/hip_bf16.h>

// ---------------------------------------------------------------------------
// out[b,i,s] = (||w[i,:]||^2 - 2*(W @ X_b)[i,s] + ||x[b,:,s]||^2) / OUT_DIM
// GEMM part runs on v_wmma_f32_16x16x32_bf16; w2/x2 computed exactly in f32.
// ---------------------------------------------------------------------------

typedef __attribute__((ext_vector_type(16))) __bf16 v16bf;
typedef __attribute__((ext_vector_type(8)))  __bf16 v8bf;
typedef __attribute__((ext_vector_type(4)))  __bf16 v4bf;
typedef __attribute__((ext_vector_type(8)))  float  v8f;

#define IN_DIM  4096
#define OUT_DIM 1024
#define SEQ     2048
#define BATCH   4

#define BM 128
#define BN 128
#define BK 32
#define LDA 40   // 32 + 8 pad bf16 elems (80 B row: 16B-aligned, conflict-free b128)
#define LDB 40

union FragBf16 { v16bf v; v8bf h[2]; };

// ------------------------- w2[i] = sum_o w[i,o]^2 ---------------------------
__global__ __launch_bounds__(256)
void w2_kernel(const float* __restrict__ w, float* __restrict__ w2) {
    __shared__ float red[256];
    const int i = blockIdx.x;
    const float* row = w + (size_t)i * OUT_DIM;
    float s = 0.f;
    for (int o = threadIdx.x; o < OUT_DIM; o += 256) {
        float v = row[o];
        s += v * v;
    }
    red[threadIdx.x] = s;
    __syncthreads();
    for (int off = 128; off > 0; off >>= 1) {
        if ((int)threadIdx.x < off) red[threadIdx.x] += red[threadIdx.x + off];
        __syncthreads();
    }
    if (threadIdx.x == 0) w2[i] = red[0];
}

// ----------------------- x2[b,s] = sum_o x[b,o,s]^2 -------------------------
__global__ __launch_bounds__(256)
void x2_kernel(const float* __restrict__ x, float* __restrict__ x2) {
    const int g = blockIdx.x * 256 + threadIdx.x;  // 0 .. B*SEQ-1
    const int b = g / SEQ;
    const int s = g % SEQ;
    const float* col = x + (size_t)b * OUT_DIM * SEQ + s;
    float a0 = 0.f, a1 = 0.f, a2 = 0.f, a3 = 0.f;
    for (int o = 0; o < OUT_DIM; o += 4) {
        float v0 = col[(size_t)(o + 0) * SEQ];
        float v1 = col[(size_t)(o + 1) * SEQ];
        float v2 = col[(size_t)(o + 2) * SEQ];
        float v3 = col[(size_t)(o + 3) * SEQ];
        a0 += v0 * v0; a1 += v1 * v1; a2 += v2 * v2; a3 += v3 * v3;
    }
    x2[g] = (a0 + a1) + (a2 + a3);
}

// ------------------------------ main GEMM -----------------------------------
__global__ __launch_bounds__(256)
void mse_gemm_kernel(const float* __restrict__ x, const float* __restrict__ w,
                     const float* __restrict__ w2, const float* __restrict__ x2,
                     float* __restrict__ out) {
    __shared__ __attribute__((aligned(16))) __bf16 sA[2][BM * LDA]; // A: [m][k]
    __shared__ __attribute__((aligned(16))) __bf16 sB[2][BN * LDB]; // X^T: [n][k]

    const int tid  = threadIdx.x;
    const int lane = tid & 31;
    const int wave = tid >> 5;       // 0..7
    const int wm   = wave & 1;       // M half of block (64 rows)
    const int wn   = wave >> 1;      // N quarter of block (32 cols)
    const int hh   = lane >> 4;      // half-wave select
    const int ml   = lane & 15;

    const int b      = blockIdx.z;
    const int blockM = blockIdx.y * BM;
    const int blockN = blockIdx.x * BN;

    const float* xb = x + (size_t)b * OUT_DIM * SEQ;

    // A staging: 2 threads per row, 16 f32 each (4x float4)
    const int arow = tid >> 1;            // 0..127
    const int acol = (tid & 1) * 16;      // 0 / 16
    const float* aPtr = w + (size_t)(blockM + arow) * OUT_DIM + acol;

    // B staging: thread covers one n, 16 k-values (coalesced across n)
    const int bn = tid & 127;             // 0..127
    const int bk = (tid >> 7) * 16;       // 0 / 16
    const float* bPtr = xb + (size_t)bk * SEQ + blockN + bn;

    float4 areg[4];
    float  breg[16];

    v8f acc[4][2];
#pragma unroll
    for (int mi = 0; mi < 4; ++mi)
#pragma unroll
        for (int ni = 0; ni < 2; ++ni)
#pragma unroll
            for (int r = 0; r < 8; ++r) acc[mi][ni][r] = 0.f;

    auto loadG = [&](int k0) {
        const float4* ap = (const float4*)(aPtr + k0);
#pragma unroll
        for (int j = 0; j < 4; ++j) areg[j] = ap[j];
        const float* bp = bPtr + (size_t)k0 * SEQ;
#pragma unroll
        for (int j = 0; j < 16; ++j) breg[j] = bp[(size_t)j * SEQ];
    };

    auto storeL = [&](int buf) {
#pragma unroll
        for (int j = 0; j < 4; ++j) {
            v4bf pk;
            pk[0] = (__bf16)areg[j].x; pk[1] = (__bf16)areg[j].y;
            pk[2] = (__bf16)areg[j].z; pk[3] = (__bf16)areg[j].w;
            *(v4bf*)&sA[buf][arow * LDA + acol + j * 4] = pk;
        }
        v8bf p0, p1;
#pragma unroll
        for (int j = 0; j < 8; ++j) { p0[j] = (__bf16)breg[j]; p1[j] = (__bf16)breg[8 + j]; }
        *(v8bf*)&sB[buf][bn * LDB + bk]     = p0;
        *(v8bf*)&sB[buf][bn * LDB + bk + 8] = p1;
    };

    auto compute = [&](int buf) {
        // B fragments: lanes 0-15 hold K=0..15, lanes 16-31 hold K=16..31; n = ml
        FragBf16 fb[2];
#pragma unroll
        for (int ni = 0; ni < 2; ++ni) {
            const __bf16* base = &sB[buf][(wn * 32 + ni * 16 + ml) * LDB + hh * 16];
            fb[ni].h[0] = *(const v8bf*)base;
            fb[ni].h[1] = *(const v8bf*)(base + 8);
        }
#pragma unroll
        for (int mi = 0; mi < 4; ++mi) {
            // A fragment: lane group 0 -> K{0..15}, group 1 -> K{16..31}
            FragBf16 fa;
            const __bf16* abase = &sA[buf][(wm * 64 + mi * 16 + ml) * LDA + hh * 16];
            fa.h[0] = *(const v8bf*)abase;
            fa.h[1] = *(const v8bf*)(abase + 8);
#pragma unroll
            for (int ni = 0; ni < 2; ++ni) {
                acc[mi][ni] = __builtin_amdgcn_wmma_f32_16x16x32_bf16(
                    /*neg_a=*/false, fa.v, /*neg_b=*/false, fb[ni].v,
                    /*c_mod=*/(short)0, acc[mi][ni],
                    /*reuse_a=*/false, /*reuse_b=*/false);
            }
        }
    };

    // ---- software-pipelined main loop (double-buffered LDS) ----
    loadG(0);
    storeL(0);
    __syncthreads();

    const int NK = OUT_DIM / BK;  // 32
    for (int kt = 0; kt < NK; ++kt) {
        const int buf = kt & 1;
        if (kt + 1 < NK) loadG((kt + 1) * BK);   // global loads fly during WMMAs
        compute(buf);
        if (kt + 1 < NK) storeL(buf ^ 1);
        __syncthreads();
    }

    // ---- fused epilogue: (w2 + x2 - 2*wx) / OUT_DIM ----
    const float inv = 1.0f / (float)OUT_DIM;
#pragma unroll
    for (int mi = 0; mi < 4; ++mi) {
        const int rowbase = blockM + wm * 64 + mi * 16 + hh * 8;
        float w2v[8];
#pragma unroll
        for (int r = 0; r < 8; ++r) w2v[r] = w2[rowbase + r];
#pragma unroll
        for (int ni = 0; ni < 2; ++ni) {
            const int scol = blockN + wn * 32 + ni * 16 + ml;
            const float x2v = x2[b * SEQ + scol];
#pragma unroll
            for (int r = 0; r < 8; ++r) {
                float val = (w2v[r] + x2v - 2.0f * acc[mi][ni][r]) * inv;
                out[((size_t)b * IN_DIM + rowbase + r) * SEQ + scol] = val;
            }
        }
    }
}

// ---------------------------------------------------------------------------
extern "C" void kernel_launch(void* const* d_in, const int* in_sizes, int n_in,
                              void* d_out, int out_size, void* d_ws, size_t ws_size,
                              hipStream_t stream) {
    const float* x = (const float*)d_in[0];   // (B, OUT_DIM, SEQ)
    const float* w = (const float*)d_in[1];   // (IN_DIM, OUT_DIM)
    float* out = (float*)d_out;               // (B, IN_DIM, SEQ)

    float* w2 = (float*)d_ws;                 // IN_DIM floats
    float* x2 = w2 + IN_DIM;                  // B*SEQ floats (total 48 KB of ws)

    w2_kernel<<<IN_DIM, 256, 0, stream>>>(w, w2);
    x2_kernel<<<(BATCH * SEQ) / 256, 256, 0, stream>>>(x, x2);

    dim3 grid(SEQ / BN, IN_DIM / BM, BATCH);
    mse_gemm_kernel<<<grid, 256, 0, stream>>>(x, w, w2, x2, out);
}
